// TemporalMSDeformAttnBase_29841432773217
// MI455X (gfx1250) — matrix operand order repaired and checked by reading, not compiled, not run on hardware
//
#include <hip/hip_runtime.h>
#include <hip/hip_bf16.h>
#include <stdint.h>

// ---------------------------------------------------------------------------
// TemporalMSDeformAttn projection head, fused for MI455X (gfx1250, wave32).
//   C = [input|query] (Mx256 f32) @ Wcat (256x1024) -> value/offsets/softmax
// bf16 WMMA (v_wmma_f32_16x16x32_bf16), f32 accumulate, fused softmax epilogue.
// Workgroup tile 64x256 (one output region per block), wave tile 32x64.
// Double-buffered A/B fragments; all inner-loop loads and epilogue stores use
// one base address + 24-bit immediate offsets (no per-access 64-bit math).
// ---------------------------------------------------------------------------

typedef __bf16 bf16x16 __attribute__((ext_vector_type(16)));
typedef float  f32x8   __attribute__((ext_vector_type(8)));

#define T_FRAMES   12
#define LEN_Q      8160
#define M_ROWS     (T_FRAMES * LEN_Q)     // 97920 rows, divisible by 64
#define K_DIM      256
#define N_TOT      1024                   // 256 val | 256 so | 256 tso | 256 aw/taw interleaved
#define LDS_STRIDE 264                    // bf16 elems per LDS row (256 + 8 pad -> conflict free)

__device__ __forceinline__ unsigned short f2bf(float f) {
  unsigned u = __float_as_uint(f);
  u += 0x7FFFu + ((u >> 16) & 1u);        // round-to-nearest-even
  return (unsigned short)(u >> 16);
}

#if defined(__has_builtin)
#if __has_builtin(__builtin_amdgcn_cvt_pk_bf16_f32)
#define HAVE_PK_BF16 1
#endif
#endif

__device__ __forceinline__ unsigned cvt2bf(float x, float y) {
#ifdef HAVE_PK_BF16
  typedef __bf16 bf16x2 __attribute__((ext_vector_type(2)));
  bf16x2 r = __builtin_amdgcn_cvt_pk_bf16_f32(x, y);   // v_cvt_pk_bf16_f32
  return __builtin_bit_cast(unsigned, r);
#else
  return (unsigned)f2bf(x) | ((unsigned)f2bf(y) << 16);
#endif
}

struct __align__(16) Frag { unsigned int u[8]; };  // 16 bf16 = one WMMA A/B operand

// Packed weight layout (fragment-linear, matches ISA 16-bit B operand striping):
//   [0,256)   W_val            [256,512) W_so        [512,768) W_tso
//   [768,1024) per-head interleave: head h -> 16 cols of W_aw then 16 of W_taw
__global__ void pack_weights(const float* __restrict__ Wval, const float* __restrict__ Wso,
                             const float* __restrict__ Wtso, const float* __restrict__ Waw,
                             const float* __restrict__ Wtaw, unsigned short* __restrict__ Bp) {
  int id   = blockIdx.x * 256 + threadIdx.x;   // 0 .. 256*1024-1
  int t    = id & 15;                          // K stream within lane
  int lane = (id >> 4) & 31;
  int ks   = (id >> 9) & 7;                    // K step (32 K per step)
  int nt   = id >> 12;                         // 16-wide packed column tile
  int K    = ks * 32 + ((lane >> 4) << 3) + (t & 7) + ((t >> 3) << 4); // lanes 0-15: K+0..7,16..23, lanes 16-31: K+8..15,24..31
  int pc   = nt * 16 + (lane & 15);            // packed column

  const float* W; int col; int Nw;
  if (pc < 256)      { W = Wval; col = pc;       Nw = 256; }
  else if (pc < 512) { W = Wso;  col = pc - 256; Nw = 256; }
  else if (pc < 768) { W = Wtso; col = pc - 512; Nw = 256; }
  else {
    int p = pc - 768, h = p >> 5, j = p & 15;
    if ((p >> 4) & 1) { W = Wtaw; col = h * 16 + j; Nw = 128; }
    else              { W = Waw;  col = h * 16 + j; Nw = 128; }
  }
  Bp[id] = f2bf(W[(size_t)K * Nw + col]);
}

__global__ void pack_bias(const float* __restrict__ bval, const float* __restrict__ bso,
                          const float* __restrict__ btso, const float* __restrict__ baw,
                          const float* __restrict__ btaw, float* __restrict__ outB) {
  int pc = blockIdx.x * 256 + threadIdx.x;
  if (pc >= N_TOT) return;
  float v;
  if (pc < 256)      v = bval[pc];
  else if (pc < 512) v = bso[pc - 256];
  else if (pc < 768) v = btso[pc - 512];
  else {
    int p = pc - 768, h = p >> 5, j = p & 15;
    v = ((p >> 4) & 1) ? btaw[h * 16 + j] : baw[h * 16 + j];
  }
  outB[pc] = v;
}

__global__ __launch_bounds__(256) void gemm_fused(
    const float* __restrict__ query, const float* __restrict__ input_flatten,
    const unsigned short* __restrict__ wsB, const float* __restrict__ wsBias,
    float* __restrict__ out) {
  __shared__ __align__(16) unsigned short sA[64 * LDS_STRIDE];

  const int tid      = threadIdx.x;
  const int lane     = tid & 31;
  const int wave     = tid >> 5;
  const int wrt      = wave >> 2;              // wave row 0..1  (32 rows each)
  const int wct      = wave & 3;               // wave col 0..3  (64 cols each)
  const int colBlock = blockIdx.x;             // 0:value 1:curr_off 2:temp_off 3:aw/taw
  const int rowBase  = blockIdx.y * 64;

  const float* Asrc = (colBlock == 0) ? input_flatten : query;

  // ---- stage A tile (64 x 256 f32) -> bf16 LDS (coalesced 16B loads) ----
  #pragma unroll
  for (int it = 0; it < 16; ++it) {
    int q  = it * 256 + tid;
    int r  = q >> 6;                           // row 0..63
    int c4 = q & 63;                           // float4 column
    const float4 f = ((const float4*)(Asrc + (size_t)(rowBase + r) * K_DIM))[c4];
    uint2 p;
    p.x = cvt2bf(f.x, f.y);
    p.y = cvt2bf(f.z, f.w);
    *(uint2*)&sA[r * LDS_STRIDE + c4 * 4] = p;
  }
  __syncthreads();

  f32x8 acc[2][4];
  #pragma unroll
  for (int i = 0; i < 2; ++i)
    #pragma unroll
    for (int j2 = 0; j2 < 4; ++j2)
      #pragma unroll
      for (int e = 0; e < 8; ++e) acc[i][j2][e] = 0.0f;

  const int half8  = (lane >> 4) << 3;         // A: lane-half K offset (0 / 8)
  const int mlocal = wrt * 32;
  const int ntBase = colBlock * 16 + wct * 4;  // packed 16-col tile index
  const int aRowOff0 = (mlocal + (lane & 15)) * LDS_STRIDE + half8;       // rt=0
  const int aRowOff1 = (mlocal + 16 + (lane & 15)) * LDS_STRIDE + half8;  // rt=1

  // Single base address for ALL 64 B-fragment loads of this wave; per-load
  // displacement ct*8192 + ks*1024 (+16) stays inside the 24-bit IOFFSET.
  const char* bBase = (const char*)wsB + (size_t)ntBase * 8192 + (size_t)lane * 32;

  // Double-buffered fragments: prefetch ks+1 while WMMAs consume ks.
  Frag af[2][2], bfr[2][4];

  #define LOAD_A(ksv, buf)                                                       \
    {                                                                            \
      int ko = (ksv) * 32;                                                       \
      *(uint4*)&af[buf][0].u[0] = *(const uint4*)&sA[aRowOff0 + ko];             \
      *(uint4*)&af[buf][0].u[4] = *(const uint4*)&sA[aRowOff0 + ko + 16];        \
      *(uint4*)&af[buf][1].u[0] = *(const uint4*)&sA[aRowOff1 + ko];             \
      *(uint4*)&af[buf][1].u[4] = *(const uint4*)&sA[aRowOff1 + ko + 16];        \
    }
  #define LOAD_B(ksv, buf)                                                       \
    {                                                                            \
      _Pragma("unroll")                                                          \
      for (int ct = 0; ct < 4; ++ct) {                                           \
        *(uint4*)&bfr[buf][ct].u[0] =                                            \
            *(const uint4*)(bBase + ct * 8192 + (ksv) * 1024);                   \
        *(uint4*)&bfr[buf][ct].u[4] =                                            \
            *(const uint4*)(bBase + ct * 8192 + (ksv) * 1024 + 16);              \
      }                                                                          \
    }

  LOAD_A(0, 0)
  LOAD_B(0, 0)

  #pragma unroll
  for (int ks = 0; ks < 8; ++ks) {
    const int cur = ks & 1, nxt = cur ^ 1;
    if (ks < 7) {                              // prefetch next K step
      LOAD_A(ks + 1, nxt)
      LOAD_B(ks + 1, nxt)
    }
    #pragma unroll
    for (int rt = 0; rt < 2; ++rt) {
      bf16x16 av = __builtin_bit_cast(bf16x16, af[cur][rt]);
      #pragma unroll
      for (int ct = 0; ct < 4; ++ct) {
        bf16x16 bv = __builtin_bit_cast(bf16x16, bfr[cur][ct]);
        acc[rt][ct] = __builtin_amdgcn_wmma_f32_16x16x32_bf16(
            false, av, false, bv, (short)0, acc[rt][ct], false, false);
      }
    }
  }
  #undef LOAD_A
  #undef LOAD_B

  // ---- epilogue: C layout is VGPR m -> row m (lanes 0-15) / m+8 (lanes 16-31), N = lane&15
  const size_t M    = (size_t)M_ROWS;
  const int    colW = colBlock * 256 + wct * 64;   // wave packed col base
  const int    j    = lane & 15;
  const int    row0 = rowBase + wrt * 32 + half8;  // first row of acc element m=0 (rt=0)

  if (colBlock < 3) {
    // value / curr_off / temp_off: bias + direct store (reshapes are contiguous)
    float* outp = out + (size_t)colBlock * M * 256;
    #pragma unroll
    for (int rt = 0; rt < 2; ++rt)
      #pragma unroll
      for (int ct = 0; ct < 4; ++ct) {
        float  bias = wsBias[colW + ct * 16 + j];
        float* p    = outp + (size_t)(row0 + rt * 16) * 256 + wct * 64 + ct * 16 + j;
        #pragma unroll
        for (int m = 0; m < 8; ++m)            // consecutive rows -> imm offsets m*1024B
          p[m * 256] = acc[rt][ct][m] + bias;
      }
  } else {
    // aw block: wave holds 2 heads; pair p -> tiles (2p)=caw, (2p+1)=taw.
    // Row-wise softmax over 32 logits across 16 lanes x 2 regs via wave32 shuffles.
    float* awc = out + 3 * M * 256;
    float* awt = awc + M * 128;
    #pragma unroll
    for (int p = 0; p < 2; ++p) {
      int   h   = wct * 2 + p;
      float bc  = wsBias[colW + p * 32 + j];
      float bt  = wsBias[colW + p * 32 + 16 + j];
      #pragma unroll
      for (int rt = 0; rt < 2; ++rt) {
        float* pc_ = awc + (size_t)(row0 + rt * 16) * 128 + h * 16 + j;
        float* pt_ = awt + (size_t)(row0 + rt * 16) * 128 + h * 16 + j;
        #pragma unroll
        for (int m = 0; m < 8; ++m) {
          float v0 = acc[rt][2 * p][m] + bc;
          float v1 = acc[rt][2 * p + 1][m] + bt;
          float mx = fmaxf(v0, v1);
          #pragma unroll
          for (int s = 1; s < 16; s <<= 1) mx = fmaxf(mx, __shfl_xor(mx, s, 32));
          float e0 = __expf(v0 - mx);
          float e1 = __expf(v1 - mx);
          float sm = e0 + e1;
          #pragma unroll
          for (int s = 1; s < 16; s <<= 1) sm += __shfl_xor(sm, s, 32);
          float inv = 1.0f / sm;
          pc_[m * 128] = e0 * inv;             // consecutive rows -> imm offsets m*512B
          pt_[m * 128] = e1 * inv;
        }
      }
    }
  }
}

extern "C" void kernel_launch(void* const* d_in, const int* in_sizes, int n_in,
                              void* d_out, int out_size, void* d_ws, size_t ws_size,
                              hipStream_t stream) {
  (void)in_sizes; (void)n_in; (void)out_size; (void)ws_size;
  const float* query  = (const float*)d_in[0];
  const float* inflat = (const float*)d_in[1];
  const float* W_val  = (const float*)d_in[2];
  const float* b_val  = (const float*)d_in[3];
  const float* W_so   = (const float*)d_in[4];
  const float* b_so   = (const float*)d_in[5];
  const float* W_aw   = (const float*)d_in[6];
  const float* b_aw   = (const float*)d_in[7];
  const float* W_tso  = (const float*)d_in[8];
  const float* b_tso  = (const float*)d_in[9];
  const float* W_taw  = (const float*)d_in[10];
  const float* b_taw  = (const float*)d_in[11];

  unsigned short* wsB   = (unsigned short*)d_ws;                       // 512 KB bf16 weights
  float*          wsBia = (float*)((char*)d_ws + (size_t)K_DIM * N_TOT * 2);  // 4 KB biases

  pack_weights<<<(K_DIM * N_TOT) / 256, 256, 0, stream>>>(W_val, W_so, W_tso, W_aw, W_taw, wsB);
  pack_bias<<<4, 256, 0, stream>>>(b_val, b_so, b_tso, b_aw, b_taw, wsBia);

  dim3 grid(4, M_ROWS / 64);   // 4 col blocks (one per output region) x 1530 row tiles
  gemm_fused<<<grid, 256, 0, stream>>>(query, inflat, wsB, wsBia, (float*)d_out);
}
